// ImitationHead_62045097558473
// MI455X (gfx1250) — matrix operand order repaired and checked by reading, not compiled, not run on hardware
//
#include <hip/hip_runtime.h>
#include <hip/hip_bf16.h>

// ---------------------------------------------------------------------------
// Types for CDNA5 WMMA (wave32): A/B = 16 bf16 per lane, C/D = 8 f32 per lane
// ---------------------------------------------------------------------------
typedef __attribute__((ext_vector_type(16))) __bf16 v16bf;
typedef __attribute__((ext_vector_type(8)))  float  v8f;

union FragBF16 { v16bf v; uint4 q[2]; unsigned int d[8]; unsigned short h[16]; };
union FragF32  { v8f   v; float f[8]; };

__device__ __forceinline__ unsigned short f2bf(float x) {
    union { float f; unsigned int u; } c; c.f = x;
    unsigned int u = c.u;
    u += 0x7FFFu + ((u >> 16) & 1u);   // round to nearest even
    return (unsigned short)(u >> 16);
}

// ---------------------------------------------------------------------------
// Kernel 0: bulk f32 -> bf16 weight conversion (one pass, packed stores).
// ---------------------------------------------------------------------------
__global__ void cvt_bf16_kernel(const float* __restrict__ src,
                                unsigned short* __restrict__ dst, int n) {
    int i = (int)(blockIdx.x * blockDim.x + threadIdx.x) * 4;
    if (i >= n) return;
    float4 v = *(const float4*)(src + i);
    uint2 r;
    r.x = (unsigned int)f2bf(v.x) | ((unsigned int)f2bf(v.y) << 16);
    r.y = (unsigned int)f2bf(v.z) | ((unsigned int)f2bf(v.w) << 16);
    *(uint2*)(dst + i) = r;
}

// ---------------------------------------------------------------------------
// Kernel 1: AdaptiveAvgPool2d(1) over 16x16 spatial, fused f32 -> bf16.
// One wave per (b,c) pair; lane reads 8 floats (2x float4); shfl reduction.
// Bandwidth-dominant kernel of the whole net (134 MB in, ~5.8us @ 23.3TB/s).
// ---------------------------------------------------------------------------
__global__ void pool_kernel(const float* __restrict__ z,
                            unsigned short* __restrict__ a0) {
    int wave = (int)((blockIdx.x * blockDim.x + threadIdx.x) >> 5); // (b*512+c)
    int lane = threadIdx.x & 31;
    const float* p = z + (size_t)wave * 256 + lane * 8;
    __builtin_prefetch(p + 4096, 0, 0);   // speculative; dropped if OOB
    float4 v0 = ((const float4*)p)[0];
    float4 v1 = ((const float4*)p)[1];
    float s = v0.x + v0.y + v0.z + v0.w + v1.x + v1.y + v1.z + v1.w;
    #pragma unroll
    for (int m = 16; m >= 1; m >>= 1) s += __shfl_xor(s, m, 32);
    if (lane == 0) a0[wave] = f2bf(s * (1.0f / 256.0f));
}

// ---------------------------------------------------------------------------
// Kernel 2: O = relu(A @ Wb^T + bias), all-bf16 operands, f32 accumulate.
//   A  : [M,K] bf16    Wb : [N,K] bf16 (pre-converted)    bias : [N] f32
// One wave per 16x16 tile via v_wmma_f32_16x16x32_bf16, K step 32.
// Per the ISA 16-bit A(16x32)/B(32x16) lane layouts, each lane's fragment
// halves are K-contiguous, so each fragment is exactly 2 global_load_b128.
// ---------------------------------------------------------------------------
__global__ void gemm_bf16_relu(const unsigned short* __restrict__ A,
                               const unsigned short* __restrict__ Wb,
                               const float* __restrict__ bias,
                               unsigned short* __restrict__ Obf,
                               float* __restrict__ Of32,
                               int M, int N, int K) {
    int lane = threadIdx.x & 31;
    int w    = (int)((blockIdx.x * blockDim.x + threadIdx.x) >> 5);
    int ntn  = N >> 4;
    int tm   = (w / ntn) << 4;
    int tn   = (w % ntn) << 4;

    int khalf = lane >> 4;          // lanes 16-31 hold the second K half
    const unsigned short* arow = A  + (size_t)(tm + (lane & 15)) * K + khalf * 8;
    const unsigned short* brow = Wb + (size_t)(tn + (lane & 15)) * K + khalf * 8;

    FragF32 acc;
    #pragma unroll
    for (int i = 0; i < 8; ++i) acc.f[i] = 0.0f;

    for (int k0 = 0; k0 < K; k0 += 32) {
        FragBF16 a, b;
        a.q[0] = *(const uint4*)(arow + k0);        // K = khalf*8 .. +7
        a.q[1] = *(const uint4*)(arow + k0 + 16);   // K = 16+khalf*8 .. +7
        b.q[0] = *(const uint4*)(brow + k0);        // K = khalf*8 .. +7
        b.q[1] = *(const uint4*)(brow + k0 + 16);   // K = 16+khalf*8 .. +7
        acc.v = __builtin_amdgcn_wmma_f32_16x16x32_bf16(
            false, a.v, false, b.v, (short)0, acc.v, false, false);
    }

    // C/D layout: VGPR i -> M = i (lanes 0-15) or 8+i (lanes 16-31), N = lane%16
    int n     = tn + (lane & 15);
    int mbase = tm + ((lane < 16) ? 0 : 8);
    float bn  = bias[n];
    #pragma unroll
    for (int i = 0; i < 8; ++i) {
        float v = acc.f[i] + bn;
        v = v > 0.0f ? v : 0.0f;
        size_t idx = (size_t)(mbase + i) * N + n;
        Of32[idx] = v;
        Obf[idx]  = f2bf(v);
    }
}

// ---------------------------------------------------------------------------
// Kernel 3: GRU recurrence (8 steps) + output MLP.
// Each block owns GRU_BB=16 batch rows (independent across blocks => no
// inter-block sync across timesteps). Gate GEMM h@W_hh^T (16x192x64) runs on
// WMMA with bf16 operands / f32 accumulation; recurrent h kept in f32.
// HS=72 bf16 row stride: 144B rows (16B aligned for ds_load_b128) and the
// 36-dword stride maps lanes 0-15 onto 16 distinct LDS banks.
// ---------------------------------------------------------------------------
#define GRU_BB 16
#define HS     72    // padded bf16/f32 row stride for K=64
#define GS     194   // padded f32 stride for the 192-wide gate matrix

// dynamic LDS layout (byte offsets all multiples of 16)
#define GRU_LDS_BYTES ((192*HS + GRU_BB*HS) * 2 + \
                       (GRU_BB*HS + GRU_BB*GS + 192*6 + 192 + 192 + GRU_BB*4 + GRU_BB*4) * 4)

__global__ void gru_kernel(const float* __restrict__ h0,      // [256,64] f32
                           const float* __restrict__ gp,      // [256,4,4]
                           const float* __restrict__ gps,     // [256]
                           const float* __restrict__ W_ih,    // [192,6]
                           const float* __restrict__ W_hh,    // [192,64]
                           const float* __restrict__ b_ih,    // [192]
                           const float* __restrict__ b_hh,    // [192]
                           const float* __restrict__ oW1, const float* __restrict__ ob1,
                           const float* __restrict__ oW2, const float* __restrict__ ob2,
                           const float* __restrict__ oW3, const float* __restrict__ ob3,
                           float* __restrict__ out) {          // [256,8,3]
    extern __shared__ __align__(16) char smem[];
    unsigned short* sWhh = (unsigned short*)smem;          // 192*HS (27648 B)
    unsigned short* sHbf = sWhh + 192 * HS;                // GRU_BB*HS (2304 B)
    float* sHf   = (float*)(sHbf + GRU_BB * HS);           // GRU_BB*HS
    float* sGh   = sHf   + GRU_BB * HS;                    // GRU_BB*GS
    float* sWih  = sGh   + GRU_BB * GS;                    // 192*6
    float* sBih  = sWih  + 192 * 6;
    float* sBhh  = sBih  + 192;
    float* sX    = sBhh  + 192;                            // GRU_BB*4
    float* sGoal = sX    + GRU_BB * 4;                     // GRU_BB*4

    const int tid  = threadIdx.x;
    const int b0   = blockIdx.x * GRU_BB;
    const int lane = tid & 31;
    const int wave = tid >> 5;

    // ---- stage weights / state into LDS ----
    for (int idx = tid; idx < 192 * 64; idx += 256) {
        int nn = idx >> 6, kk = idx & 63;
        sWhh[nn * HS + kk] = f2bf(W_hh[idx]);
    }
    for (int idx = tid; idx < 192 * 6; idx += 256) sWih[idx] = W_ih[idx];
    for (int idx = tid; idx < 192; idx += 256) { sBih[idx] = b_ih[idx]; sBhh[idx] = b_hh[idx]; }
    for (int idx = tid; idx < GRU_BB * 64; idx += 256) {
        int bb = idx >> 6, jj = idx & 63;
        float h = h0[(size_t)(b0 + bb) * 64 + jj];
        sHf [bb * HS + jj] = h;
        sHbf[bb * HS + jj] = f2bf(h);
    }
    if (tid < GRU_BB) {
        int b = b0 + tid;
        sGoal[tid * 4 + 0] = gp[b * 16 + 3];   // goal_point[b,0,3]
        sGoal[tid * 4 + 1] = gp[b * 16 + 7];   // goal_point[b,1,3]
        sGoal[tid * 4 + 2] = gps[b];
        sX[tid * 4 + 0] = 0.f; sX[tid * 4 + 1] = 0.f; sX[tid * 4 + 2] = 0.f;
    }
    __syncthreads();

    for (int t = 0; t < 8; ++t) {
        // ---- gate GEMM: sGh[16,192] = sHbf[16,64] @ sWhh^T via WMMA ----
        // tiles: 1 (m) x 12 (n); waves take tiles wave, wave+8
        for (int tile = wave; tile < 12; tile += 8) {
            int tn = tile << 4;
            FragF32 acc;
            #pragma unroll
            for (int i = 0; i < 8; ++i) acc.f[i] = 0.0f;
            int khalf = lane >> 4;
            const unsigned short* arow = sHbf + (lane & 15) * HS + khalf * 8;
            const unsigned short* brow = sWhh + (tn + (lane & 15)) * HS + khalf * 8;
            #pragma unroll
            for (int k0 = 0; k0 < 64; k0 += 32) {
                FragBF16 a, b;
                a.q[0] = *(const uint4*)(arow + k0);
                a.q[1] = *(const uint4*)(arow + k0 + 16);
                b.q[0] = *(const uint4*)(brow + k0);
                b.q[1] = *(const uint4*)(brow + k0 + 16);
                acc.v = __builtin_amdgcn_wmma_f32_16x16x32_bf16(
                    false, a.v, false, b.v, (short)0, acc.v, false, false);
            }
            int n  = tn + (lane & 15);
            int mb = (lane < 16) ? 0 : 8;
            #pragma unroll
            for (int i = 0; i < 8; ++i) sGh[(mb + i) * GS + n] = acc.f[i];
        }
        __syncthreads();

        // ---- elementwise gate math: 16*64 elements, 4 per thread ----
        for (int e = tid; e < GRU_BB * 64; e += 256) {
            int bb = e >> 6, j = e & 63;
            float x0 = sX[bb*4+0], x1 = sX[bb*4+1], x2 = sX[bb*4+2];
            float g0 = sGoal[bb*4+0], g1 = sGoal[bb*4+1], g2 = sGoal[bb*4+2];
            float gi[3], gh[3];
            #pragma unroll
            for (int q = 0; q < 3; ++q) {
                int row = q * 64 + j;
                const float* wr = sWih + row * 6;
                gi[q] = sBih[row] + wr[0]*x0 + wr[1]*x1 + wr[2]*x2
                                  + wr[3]*g0 + wr[4]*g1 + wr[5]*g2;
                gh[q] = sBhh[row] + sGh[bb * GS + row];
            }
            float r  = 1.0f / (1.0f + __expf(-(gi[0] + gh[0])));
            float zg = 1.0f / (1.0f + __expf(-(gi[1] + gh[1])));
            float nv = tanhf(gi[2] + r * gh[2]);
            float hn = (1.0f - zg) * nv + zg * sHf[bb * HS + j];
            sHf [bb * HS + j] = hn;
            sHbf[bb * HS + j] = f2bf(hn);
        }
        __syncthreads();

        // ---- output MLP 64->4(relu)->4->3, x += dx ; one thread per row ----
        if (tid < GRU_BB) {
            int bb = tid;
            float d1[4];
            #pragma unroll
            for (int o = 0; o < 4; ++o) {
                float s = ob1[o];
                for (int k = 0; k < 64; ++k) s += oW1[o * 64 + k] * sHf[bb * HS + k];
                d1[o] = s > 0.0f ? s : 0.0f;
            }
            float d2[4];
            #pragma unroll
            for (int o = 0; o < 4; ++o) {
                float s = ob2[o];
                #pragma unroll
                for (int k = 0; k < 4; ++k) s += oW2[o * 4 + k] * d1[k];
                d2[o] = s;
            }
            float d3[3];
            #pragma unroll
            for (int o = 0; o < 3; ++o) {
                float s = ob3[o];
                #pragma unroll
                for (int k = 0; k < 4; ++k) s += oW3[o * 4 + k] * d2[k];
                d3[o] = s;
            }
            float nx0 = sX[bb*4+0] + d3[0];
            float nx1 = sX[bb*4+1] + d3[1];
            float nx2 = sX[bb*4+2] + d3[2];
            sX[bb*4+0] = nx0; sX[bb*4+1] = nx1; sX[bb*4+2] = nx2;
            size_t o = (size_t)(b0 + bb) * 24 + t * 3;
            out[o+0] = nx0; out[o+1] = nx1; out[o+2] = nx2;
        }
        __syncthreads();
    }
}

// ---------------------------------------------------------------------------
// Host-side orchestration
// ---------------------------------------------------------------------------
extern "C" void kernel_launch(void* const* d_in, const int* in_sizes, int n_in,
                              void* d_out, int out_size, void* d_ws, size_t ws_size,
                              hipStream_t stream) {
    const float* z    = (const float*)d_in[0];
    const float* gp   = (const float*)d_in[1];
    // d_in[2], d_in[3]: gt_trajectory / speed — unused in eval forward
    const float* gps  = (const float*)d_in[4];
    const float* jW1  = (const float*)d_in[5];  const float* jb1 = (const float*)d_in[6];
    const float* jW2  = (const float*)d_in[7];  const float* jb2 = (const float*)d_in[8];
    const float* jW3  = (const float*)d_in[9];  const float* jb3 = (const float*)d_in[10];
    const float* jW4  = (const float*)d_in[11]; const float* jb4 = (const float*)d_in[12];
    const float* W_ih = (const float*)d_in[13]; const float* W_hh = (const float*)d_in[14];
    const float* b_ih = (const float*)d_in[15]; const float* b_hh = (const float*)d_in[16];
    const float* oW1  = (const float*)d_in[17]; const float* ob1 = (const float*)d_in[18];
    const float* oW2  = (const float*)d_in[19]; const float* ob2 = (const float*)d_in[20];
    const float* oW3  = (const float*)d_in[21]; const float* ob3 = (const float*)d_in[22];
    float* out = (float*)d_out;

    const int B = 256;
    // --- workspace carve (256B-aligned slabs) ---
    char* ws = (char*)d_ws;
    size_t off = 0;
    auto carve = [&](size_t bytes) {
        char* p = ws + off;
        off += (bytes + 255) & ~(size_t)255;
        return p;
    };
    unsigned short* A0  = (unsigned short*)carve((size_t)B * 512 * 2);
    unsigned short* A1b = (unsigned short*)carve((size_t)B * 512 * 2);
    float*          A1f = (float*)         carve((size_t)B * 512 * 4);
    unsigned short* A2b = (unsigned short*)carve((size_t)B * 256 * 2);
    float*          A2f = (float*)         carve((size_t)B * 256 * 4);
    unsigned short* A3b = (unsigned short*)carve((size_t)B * 128 * 2);
    float*          A3f = (float*)         carve((size_t)B * 128 * 4);
    unsigned short* A4b = (unsigned short*)carve((size_t)B * 64 * 2);
    float*          A4f = (float*)         carve((size_t)B * 64 * 4);
    unsigned short* W1b = (unsigned short*)carve((size_t)512 * 512 * 2);
    unsigned short* W2b = (unsigned short*)carve((size_t)256 * 512 * 2);
    unsigned short* W3b = (unsigned short*)carve((size_t)128 * 256 * 2);
    unsigned short* W4b = (unsigned short*)carve((size_t)64  * 128 * 2);

    // 0) one-time weight conversion to bf16 (packed b128/b64 loads+stores)
    auto cvt = [&](const float* s, unsigned short* d, int n) {
        cvt_bf16_kernel<<<(n / 4 + 255) / 256, 256, 0, stream>>>(s, d, n);
    };
    cvt(jW1, W1b, 512 * 512);
    cvt(jW2, W2b, 256 * 512);
    cvt(jW3, W3b, 128 * 256);
    cvt(jW4, W4b, 64 * 128);

    // 1) pooling: 256*512 waves, 8 waves/block
    pool_kernel<<<(B * 512) / 8, 256, 0, stream>>>(z, A0);

    // 2) join MLP: one wave per 16x16 tile, 8 waves/block (tile counts % 8 == 0)
    auto gemm = [&](const unsigned short* A, const unsigned short* W, const float* bias,
                    unsigned short* Ob, float* Of, int M, int N, int K) {
        int waves = (M / 16) * (N / 16);
        gemm_bf16_relu<<<waves / 8, 256, 0, stream>>>(A, W, bias, Ob, Of, M, N, K);
    };
    gemm(A0,  W1b, jb1, A1b, A1f, B, 512, 512);
    gemm(A1b, W2b, jb2, A2b, A2f, B, 256, 512);
    gemm(A2b, W3b, jb3, A3b, A3f, B, 128, 256);
    gemm(A3b, W4b, jb4, A4b, A4f, B,  64, 128);

    // 3) GRU + head: 16 blocks x 16 batch rows, dynamic LDS (~52 KB/WG)
    gru_kernel<<<B / GRU_BB, 256, GRU_LDS_BYTES, stream>>>(
        A4f, gp, gps, W_ih, W_hh, b_ih, b_hh,
        oW1, ob1, oW2, ob2, oW3, ob3, out);

    (void)in_sizes; (void)n_in; (void)out_size; (void)ws_size;
}
